// Self_Attention_25563645346060
// MI455X (gfx1250) — matrix operand order repaired and checked by reading, not compiled, not run on hardware
//
#include <hip/hip_runtime.h>

// Self-attention for B=32, d=C=384, H=W=32 (S=1024), fp32 in/out.
// All three GEMMs run on v_wmma_f32_16x16x32_bf16 with LDS-staged 128x128
// block tiles fed by double-buffered global_load_async_to_lds_b128 (ASYNCcnt).

typedef __attribute__((ext_vector_type(16))) __bf16 v16bf;
typedef __attribute__((ext_vector_type(8)))  float  v8f;

#define B_   32
#define D_   384
#define D3_  1152
#define S_   1024

#define ROWP 40                      // padded LDS row pitch in halves (80 B)
#define PSH  (128 * ROWP)            // panel size in halves
#define PSB  (PSH * 2)               // panel size in bytes (10240)

__device__ __forceinline__ unsigned short f2bf(float f) {
  unsigned int u = __float_as_uint(f);
  u += 0x7FFFu + ((u >> 16) & 1u);   // round-to-nearest-even
  return (unsigned short)(u >> 16);
}

union V16U { v16bf v; uint4 q[2]; };
union H8U  { unsigned short h[8]; uint4 q; };

__device__ __forceinline__ v8f wmma_bf16(v16bf a, v16bf b, v8f c) {
  return __builtin_amdgcn_wmma_f32_16x16x32_bf16(false, a, false, b, (short)0, c, false, false);
}

// ---- LDS tile readers (layouts per CDNA5 ISA §7.12.2) -----------
// A-matrix 16x32 bf16 from padded panel: lane = row, chunks at kb and kb+16
__device__ __forceinline__ v16bf lds_tileA(const unsigned short* sp, int m0) {
  int lane = threadIdx.x & 31;
  const unsigned short* row = sp + (m0 + (lane & 15)) * ROWP + ((lane >> 4) << 3);
  V16U u;
  u.q[0] = *(const uint4*)(row);
  u.q[1] = *(const uint4*)(row + 16);
  return u.v;
}
// B-matrix 32x16 bf16 (Bt row-major in panel): lane = column, chunks at kb and kb+16
__device__ __forceinline__ v16bf lds_tileB(const unsigned short* sp, int n0) {
  int lane = threadIdx.x & 31;
  const unsigned short* row = sp + (n0 + (lane & 15)) * ROWP + ((lane >> 4) << 3);
  V16U u;
  u.q[0] = *(const uint4*)(row);
  u.q[1] = *(const uint4*)(row + 16);
  return u.v;
}

// ---- async copy of one 128x32-half panel to LDS (256 threads, 2 B128 each) -
__device__ __forceinline__ void panel_async(const unsigned short* __restrict__ g0,
                                            int ld, int r0, int k0,
                                            unsigned ldsByteBase) {
  int tid = threadIdx.x;
#pragma unroll
  for (int i = 0; i < 2; ++i) {
    int c    = tid * 2 + i;            // 0..511 chunks of 16B
    int row  = c >> 2;
    int colh = (c & 3) * 8;            // halves
    const unsigned short* g = g0 + (size_t)(r0 + row) * ld + k0 + colh;
    unsigned loff = ldsByteBase + row * (ROWP * 2) + colh * 2;
    unsigned long long ga = (unsigned long long)g;
    asm volatile("global_load_async_to_lds_b128 %0, %1, off"
                 :: "v"(loff), "v"(ga) : "memory");
  }
}

// ---- 128x128 block GEMM mainloop: D += A(128xK) * Bt(128xK)^T --------------
__device__ __forceinline__ void gemm128x128(const unsigned short* __restrict__ Ab, int ldA, int m0blk,
                                            const unsigned short* __restrict__ Bb, int ldB, int n0blk,
                                            int kTotal, unsigned short* sm,
                                            v8f (&acc)[2][4], int mw, int nw) {
  const unsigned smbase = (unsigned)(unsigned long long)sm;  // LDS byte offset
  const int nk = kTotal / 32;
  panel_async(Ab, ldA, m0blk, 0, smbase);
  panel_async(Bb, ldB, n0blk, 0, smbase + PSB);
  for (int i = 0; i < nk; ++i) {
    int cur = i & 1;
    asm volatile("s_wait_asynccnt 0" ::: "memory");
    __syncthreads();                                   // panels[cur] visible to all
    if (i + 1 < nk) {
      int nxt = cur ^ 1;
      panel_async(Ab, ldA, m0blk, (i + 1) * 32, smbase + nxt * 2 * PSB);
      panel_async(Bb, ldB, n0blk, (i + 1) * 32, smbase + nxt * 2 * PSB + PSB);
    }
    const unsigned short* pA = sm + cur * 2 * PSH;
    const unsigned short* pB = pA + PSH;
    v16bf a0 = lds_tileA(pA, mw);
    v16bf a1 = lds_tileA(pA, mw + 16);
#pragma unroll
    for (int t = 0; t < 4; ++t) {
      v16bf bt = lds_tileB(pB, nw + t * 16);
      acc[0][t] = wmma_bf16(a0, bt, acc[0][t]);
      acc[1][t] = wmma_bf16(a1, bt, acc[1][t]);
    }
    __syncthreads();                                   // done reading panels[cur]
  }
}

// ---- Kernel 1: x[B,C,S] -> xs[B,S,C] bf16, + positional encoding -----------
__global__ __launch_bounds__(256) void k_prep_x(const float* __restrict__ x,
                                                unsigned short* __restrict__ xs) {
  __shared__ float tile[32][33];
  int b  = blockIdx.z;
  int s0 = blockIdx.x * 32, c0 = blockIdx.y * 32;
  int tx = threadIdx.x, ty = threadIdx.y;              // (32,8)
  const float* xb = x + (size_t)b * D_ * S_;
#pragma unroll
  for (int j = 0; j < 4; ++j) {
    int cl = ty + j * 8;
    tile[cl][tx] = xb[(size_t)(c0 + cl) * S_ + s0 + tx];
  }
  __syncthreads();
#pragma unroll
  for (int j = 0; j < 4; ++j) {
    int s = s0 + ty + j * 8;
    int c = c0 + tx;
    float div = expf(-9.210340371976184f * (float)(c & ~1) * (1.0f / (float)D_));
    float ang = (float)s * div;
    float pe  = (c & 1) ? cosf(ang) : sinf(ang);
    xs[(size_t)(b * S_ + s) * D_ + c] = f2bf(tile[tx][ty + j * 8] + pe);
  }
}

// ---- Kernel 2: W[d,3d] f32 -> Wt[3d,d] bf16 --------------------------------
__global__ __launch_bounds__(256) void k_prep_w(const float* __restrict__ W,
                                                unsigned short* __restrict__ Wt) {
  __shared__ float tile[32][33];
  int n0 = blockIdx.x * 32;
  int k0 = blockIdx.y * 32;
  int tx = threadIdx.x, ty = threadIdx.y;
#pragma unroll
  for (int j = 0; j < 4; ++j)
    tile[ty + j * 8][tx] = W[(size_t)(k0 + ty + j * 8) * D3_ + n0 + tx];
  __syncthreads();
#pragma unroll
  for (int j = 0; j < 4; ++j)
    Wt[(size_t)(n0 + ty + j * 8) * D_ + k0 + tx] = f2bf(tile[tx][ty + j * 8]);
}

// ---- Kernel 3: QKV projection; Q scaled; V stored transposed ---------------
__global__ __launch_bounds__(256) void k_qkv(const unsigned short* __restrict__ xs,
                                             const unsigned short* __restrict__ Wt,
                                             const float* __restrict__ bias,
                                             unsigned short* __restrict__ Q,
                                             unsigned short* __restrict__ K,
                                             unsigned short* __restrict__ Vt) {
  __shared__ unsigned short sm[4 * PSH];
  int b     = blockIdx.z;
  int m0blk = blockIdx.y * 128;
  int n0blk = blockIdx.x * 128;
  int w  = threadIdx.x >> 5;
  int mw = (w & 3) * 32, nw = (w >> 2) * 64;
  v8f acc[2][4] = {};
  gemm128x128(xs + (size_t)b * S_ * D_, D_, m0blk, Wt, D_, n0blk, D_, sm, acc, mw, nw);

  int lane = threadIdx.x & 31;
  int nl = lane & 15, hi = lane >> 4;
  int seg = n0blk / D_;                 // 0=Q, 1=K, 2=V (128 | 384)
#pragma unroll
  for (int mi = 0; mi < 2; ++mi)
#pragma unroll
    for (int t = 0; t < 4; ++t) {
      int n     = n0blk + nw + t * 16 + nl;
      int mbase = m0blk + mw + mi * 16 + 8 * hi;
      float bv  = bias[n];
      if (seg == 0) {
#pragma unroll
        for (int r = 0; r < 8; ++r)
          Q[(size_t)(b * S_ + mbase + r) * D_ + n] =
              f2bf((acc[mi][t][r] + bv) * 0.05103103630798287f);
      } else if (seg == 1) {
#pragma unroll
        for (int r = 0; r < 8; ++r)
          K[(size_t)(b * S_ + mbase + r) * D_ + (n - D_)] = f2bf(acc[mi][t][r] + bv);
      } else {
        H8U u;
#pragma unroll
        for (int r = 0; r < 8; ++r) u.h[r] = f2bf(acc[mi][t][r] + bv);
        *(uint4*)&Vt[(size_t)(b * D_ + (n - 2 * D_)) * S_ + mbase] = u.q;
      }
    }
}

// ---- Kernel 4: scores = Q K^T (Q pre-scaled), f32 to workspace -------------
__global__ __launch_bounds__(256) void k_scores(const unsigned short* __restrict__ Q,
                                                const unsigned short* __restrict__ K,
                                                float* __restrict__ Sc) {
  __shared__ unsigned short sm[4 * PSH];
  int b     = blockIdx.z;
  int m0blk = blockIdx.y * 128;
  int n0blk = blockIdx.x * 128;
  int w  = threadIdx.x >> 5;
  int mw = (w & 3) * 32, nw = (w >> 2) * 64;
  v8f acc[2][4] = {};
  gemm128x128(Q + (size_t)b * S_ * D_, D_, m0blk,
              K + (size_t)b * S_ * D_, D_, n0blk, D_, sm, acc, mw, nw);

  int lane = threadIdx.x & 31;
  int nl = lane & 15, hi = lane >> 4;
#pragma unroll
  for (int mi = 0; mi < 2; ++mi)
#pragma unroll
    for (int t = 0; t < 4; ++t) {
      int n     = n0blk + nw + t * 16 + nl;
      int mbase = m0blk + mw + mi * 16 + 8 * hi;
#pragma unroll
      for (int r = 0; r < 8; ++r)
        Sc[(size_t)(b * S_ + mbase + r) * S_ + n] = acc[mi][t][r];
    }
}

// ---- Kernel 5: in-place row softmax; rewrites row as bf16 probabilities ----
__global__ __launch_bounds__(256) void k_softmax(float* __restrict__ Sc) {
  __shared__ float red[256];
  int row = blockIdx.x;                       // b*S + q
  float* rp = Sc + (size_t)row * S_;
  int t = threadIdx.x;
  float v0 = rp[t], v1 = rp[t + 256], v2 = rp[t + 512], v3 = rp[t + 768];
  float mx = fmaxf(fmaxf(v0, v1), fmaxf(v2, v3));
  red[t] = mx; __syncthreads();
  for (int s = 128; s > 0; s >>= 1) {
    if (t < s) red[t] = fmaxf(red[t], red[t + s]);
    __syncthreads();
  }
  mx = red[0]; __syncthreads();
  float e0 = expf(v0 - mx), e1 = expf(v1 - mx), e2 = expf(v2 - mx), e3 = expf(v3 - mx);
  red[t] = e0 + e1 + e2 + e3; __syncthreads();
  for (int s = 128; s > 0; s >>= 1) {
    if (t < s) red[t] += red[t + s];
    __syncthreads();
  }
  float inv = 1.0f / red[0];
  unsigned short* pp = (unsigned short*)rp;   // bf16 row, pitch stays 2048 elems
  pp[t]       = f2bf(e0 * inv);
  pp[t + 256] = f2bf(e1 * inv);
  pp[t + 512] = f2bf(e2 * inv);
  pp[t + 768] = f2bf(e3 * inv);
}

// ---- Kernel 6: out = P @ V, written directly as [B,C,S] --------------------
__global__ __launch_bounds__(256) void k_out(const float* __restrict__ Sc,
                                             const unsigned short* __restrict__ Vt,
                                             float* __restrict__ out) {
  __shared__ unsigned short sm[4 * PSH];
  int b     = blockIdx.z;
  int m0blk = blockIdx.y * 128;   // query rows
  int n0blk = blockIdx.x * 128;   // channels
  int w  = threadIdx.x >> 5;
  int mw = (w & 3) * 32, nw = (w >> 2) * 64;
  const unsigned short* Pb = (const unsigned short*)Sc + (size_t)b * S_ * (2 * S_);
  v8f acc[2][4] = {};
  gemm128x128(Pb, 2 * S_, m0blk, Vt + (size_t)b * D_ * S_, S_, n0blk, S_, sm, acc, mw, nw);

  int lane = threadIdx.x & 31;
  int nl = lane & 15, hi = lane >> 4;
#pragma unroll
  for (int mi = 0; mi < 2; ++mi)
#pragma unroll
    for (int t = 0; t < 4; ++t) {
      int n     = n0blk + nw + t * 16 + nl;
      int mbase = m0blk + mw + mi * 16 + 8 * hi;
      float* op = out + (size_t)(b * D_ + n) * S_ + mbase;   // 8 contiguous M
      *(float4*)(op)     = make_float4(acc[mi][t][0], acc[mi][t][1], acc[mi][t][2], acc[mi][t][3]);
      *(float4*)(op + 4) = make_float4(acc[mi][t][4], acc[mi][t][5], acc[mi][t][6], acc[mi][t][7]);
    }
}

// ---- Host launcher ---------------------------------------------------------
extern "C" void kernel_launch(void* const* d_in, const int* in_sizes, int n_in,
                              void* d_out, int out_size, void* d_ws, size_t ws_size,
                              hipStream_t stream) {
  const float* x    = (const float*)d_in[0];
  const float* W    = (const float*)d_in[1];
  const float* bias = (const float*)d_in[2];
  float* out        = (float*)d_out;

  char* ws = (char*)d_ws;
  const size_t SZ_XS = (size_t)B_ * S_ * D_ * 2;   // 25,165,824
  const size_t SZ_WT = (size_t)D3_ * D_ * 2;       //    884,736
  unsigned short* xs = (unsigned short*)(ws);
  unsigned short* Wt = (unsigned short*)(ws + SZ_XS);
  unsigned short* Q  = (unsigned short*)(ws + SZ_XS + SZ_WT);
  unsigned short* K  = (unsigned short*)(ws + SZ_XS + SZ_WT + SZ_XS);
  unsigned short* Vt = (unsigned short*)(ws + SZ_XS + SZ_WT + 2 * SZ_XS);
  float*          Sc = (float*)(ws + SZ_XS + SZ_WT + 3 * SZ_XS);  // B*S*S f32

  k_prep_x <<<dim3(S_ / 32, D_ / 32, B_), dim3(32, 8), 0, stream>>>(x, xs);
  k_prep_w <<<dim3(D3_ / 32, D_ / 32, 1), dim3(32, 8), 0, stream>>>(W, Wt);
  k_qkv    <<<dim3(D3_ / 128, S_ / 128, B_), 256, 0, stream>>>(xs, Wt, bias, Q, K, Vt);
  k_scores <<<dim3(S_ / 128, S_ / 128, B_), 256, 0, stream>>>(Q, K, Sc);
  k_softmax<<<dim3(B_ * S_), 256, 0, stream>>>(Sc);
  k_out    <<<dim3(D_ / 128, S_ / 128, B_), 256, 0, stream>>>(Sc, Vt, out);
}
